// gcn_78469052498537
// MI455X (gfx1250) — matrix operand order repaired and checked by reading, not compiled, not run on hardware
//
#include <hip/hip_runtime.h>
#include <hip/hip_bf16.h>

// ---------------------------------------------------------------------------
// GCN forward for MI455X (gfx1250, wave32).
// GEMMs: v_wmma_f32_16x16x32_bf16 with hi/lo bf16 split of fp32 operands
// (Ahi*Bhi + Ahi*Blo + Alo*Bhi, fp32 accumulate) for ~fp32 accuracy.
// Block tile 64x64, 8 waves, each wave owns a 16x32 output slab (6 WMMA/chunk).
// Weight tiles staged with global_load_async_to_lds_b128 (ASYNCcnt path) from
// pre-transposed bf16 hi/lo planes; A tiles staged sync (f32->bf16 split).
// ---------------------------------------------------------------------------

typedef __bf16 v16bf __attribute__((ext_vector_type(16)));
typedef float  v8f   __attribute__((ext_vector_type(8)));
static_assert(sizeof(v16bf) == 32, "v16bf must be 32 bytes");

#define LDSS 40   // LDS row stride (elements): 80B rows -> b128-aligned segments

// convert 2 floats to packed bf16 hi-parts and lo-parts (split precision)
__device__ __forceinline__ void cvt2(float a, float b, unsigned& h, unsigned& l) {
    __bf16 ha = (__bf16)a, hb = (__bf16)b;
    __bf16 la = (__bf16)(a - (float)ha), lb = (__bf16)(b - (float)hb);
    union { __bf16 x[2]; unsigned u; } uh, ul;
    uh.x[0] = ha; uh.x[1] = hb;
    ul.x[0] = la; ul.x[1] = lb;
    h = uh.u; l = ul.u;
}

union U4 { unsigned u[4]; uint4 q; };
union F16 { unsigned u[8]; v16bf v; };

// ---------------------------------------------------------------------------
// Tiled GEMM:  C[M,N] = A[M,K] @ B (+ bias), B given pre-transposed as
// bf16 planes WtH/WtL with layout [n][k] (N x K, row stride K).
// 256 threads = 8 waves (4 M-tiles x 2 N-slabs); K mult of 32, N mult of 64.
// ---------------------------------------------------------------------------
__global__ __launch_bounds__(256) void k_gemm_wmma(
    const float* __restrict__ A,
    const __bf16* __restrict__ WtH, const __bf16* __restrict__ WtL,
    const float* __restrict__ bias, float* __restrict__ C,
    int M, int K, int N)
{
    __shared__ __bf16 Ah [64 * LDSS];   // A tile 64(M) x 32(K), hi
    __shared__ __bf16 Al [64 * LDSS];   // lo
    __shared__ __bf16 BtH[64 * LDSS];   // W tile [n][k], 64(N) x 32(K), hi
    __shared__ __bf16 BtL[64 * LDSS];   // lo

    const int tid  = threadIdx.x;
    const int lane = tid & 31;
    const int wave = tid >> 5;
    const int half = lane >> 4;      // 0: lanes 0-15, 1: lanes 16-31
    const int l16  = lane & 15;

    const int m0 = blockIdx.x * 64;
    const int n0 = blockIdx.y * 64;
    const int tm = (wave & 3) * 16;     // wave tile row inside 64
    const int tn = (wave >> 2) * 32;    // wave slab col inside 64 (two 16-wide tiles)

    // A staging coordinates (sync path)
    const int ar  = tid >> 2;           // A row   (64 rows, 4 threads/row)
    const int ac0 = (tid & 3) * 8;      // A col0  (8 K-elems per thread)
    // B staging coordinates (async path): 16B segment per thread per plane
    const int brow = tid >> 2;          // 0..63 (n within tile)
    const int bseg = tid & 3;           // 4 x 16B segments per 64B row

    const int gra  = m0 + ar;
    const int grac = (gra < M - 1) ? gra : (M - 1);   // clamped row for safe load
    const float* aptr = A + (size_t)grac * K + ac0;

    const unsigned ldsBH = (unsigned)(size_t)&BtH[brow * LDSS] + bseg * 16;
    const unsigned ldsBL = (unsigned)(size_t)&BtL[brow * LDSS] + bseg * 16;
    const __bf16* gBH = WtH + (size_t)(n0 + brow) * K + bseg * 8;
    const __bf16* gBL = WtL + (size_t)(n0 + brow) * K + bseg * 8;

    v8f acc0 = {}, acc1 = {};

    for (int k0 = 0; k0 < K; k0 += 32) {
        // ---- kick off async B tile copy (bf16 planes, byte-exact) ----------
        asm volatile("global_load_async_to_lds_b128 %0, %1, off"
                     :: "v"(ldsBH), "v"((unsigned long long)(size_t)(gBH + k0))
                     : "memory");
        asm volatile("global_load_async_to_lds_b128 %0, %1, off"
                     :: "v"(ldsBL), "v"((unsigned long long)(size_t)(gBL + k0))
                     : "memory");

        // ---- stage A 64x32 sync: 2x global_load_b128, f32 -> bf16 hi/lo ----
        float4 va0 = *(const float4*)(aptr + k0);
        float4 va1 = *(const float4*)(aptr + k0 + 4);
        if (gra >= M) { va0 = float4{0,0,0,0}; va1 = float4{0,0,0,0}; }
        if (k0 + 32 < K) __builtin_prefetch(aptr + k0 + 32, 0, 0);
        {
            U4 uh, ul;
            cvt2(va0.x, va0.y, uh.u[0], ul.u[0]);
            cvt2(va0.z, va0.w, uh.u[1], ul.u[1]);
            cvt2(va1.x, va1.y, uh.u[2], ul.u[2]);
            cvt2(va1.z, va1.w, uh.u[3], ul.u[3]);
            *(uint4*)&Ah[ar * LDSS + ac0] = uh.q;    // ds_store_b128
            *(uint4*)&Al[ar * LDSS + ac0] = ul.q;
        }

        // async copies must land before anyone reads the B tile
        asm volatile("s_wait_asynccnt 0x0" ::: "memory");
        __syncthreads();

        // ---- fragments per ISA 7.12.2 (wave32, 16-bit operands) ----
        F16 fah, fal, fbh0, fbl0, fbh1, fbl1;
        const int arow = (tm + l16) * LDSS;
        const int b0   = (tn + l16) * LDSS;
        const int b1   = (tn + 16 + l16) * LDSS;
        #pragma unroll
        for (int v = 0; v < 8; ++v) {
            // A 16x32: VGPR v holds K = 2*(v&3) + 16*(v>>2) + 8*half, +{0,1}
            int ka = ((v & 3) << 1) + ((v >> 2) << 4) + (half << 3);
            fah.u[v] = *(const unsigned*)&Ah[arow + ka];
            fal.u[v] = *(const unsigned*)&Al[arow + ka];
            // B 32x16: VGPR v holds K = 2*(v&3) + 16*(v>>2) + 8*half, +{0,1}; N = lane%16
            int kb = ((v & 3) << 1) + ((v >> 2) << 4) + (half << 3);
            fbh0.u[v] = *(const unsigned*)&BtH[b0 + kb];
            fbl0.u[v] = *(const unsigned*)&BtL[b0 + kb];
            fbh1.u[v] = *(const unsigned*)&BtH[b1 + kb];
            fbl1.u[v] = *(const unsigned*)&BtL[b1 + kb];
        }

        // split-precision accumulate, A fragments reused across both N tiles
        acc0 = __builtin_amdgcn_wmma_f32_16x16x32_bf16(false, fal.v, false, fbh0.v,
                                                       (short)0, acc0, false, false);
        acc0 = __builtin_amdgcn_wmma_f32_16x16x32_bf16(false, fah.v, false, fbl0.v,
                                                       (short)0, acc0, false, false);
        acc0 = __builtin_amdgcn_wmma_f32_16x16x32_bf16(false, fah.v, false, fbh0.v,
                                                       (short)0, acc0, false, false);
        acc1 = __builtin_amdgcn_wmma_f32_16x16x32_bf16(false, fal.v, false, fbh1.v,
                                                       (short)0, acc1, false, false);
        acc1 = __builtin_amdgcn_wmma_f32_16x16x32_bf16(false, fah.v, false, fbl1.v,
                                                       (short)0, acc1, false, false);
        acc1 = __builtin_amdgcn_wmma_f32_16x16x32_bf16(false, fah.v, false, fbh1.v,
                                                       (short)0, acc1, false, false);
        __syncthreads();
    }

    // D layout: VGPR r -> M = r + 8*half ; N = lane%16
    const int col0 = n0 + tn + l16;
    const int col1 = col0 + 16;
    const float ba0 = bias ? bias[col0] : 0.0f;
    const float ba1 = bias ? bias[col1] : 0.0f;
    #pragma unroll
    for (int r = 0; r < 8; ++r) {
        int row = m0 + tm + (half << 3) + r;
        if (row < M) {
            C[(size_t)row * N + col0] = acc0[r] + ba0;
            C[(size_t)row * N + col1] = acc1[r] + ba1;
        }
    }
}

// ---------------------------------------------------------------------------
// one-time weight prep: W[K,N] f32 -> transposed bf16 planes WtH/WtL [N,K]
// ---------------------------------------------------------------------------
__global__ void k_wprep(const float* __restrict__ W,
                        __bf16* __restrict__ WtH, __bf16* __restrict__ WtL,
                        int K, int N) {
    int idx = blockIdx.x * 256 + threadIdx.x;
    if (idx < K * N) {
        int k = idx / N, n = idx - k * N;
        float v = W[idx];
        __bf16 h = (__bf16)v;
        __bf16 l = (__bf16)(v - (float)h);
        WtH[(size_t)n * K + k] = h;
        WtL[(size_t)n * K + k] = l;
    }
}

// ---------------------------------------------------------------------------
// helpers
// ---------------------------------------------------------------------------
__global__ void k_zero(float* __restrict__ p, int n) {
    int i = blockIdx.x * 256 + threadIdx.x;
    if (i < n) p[i] = 0.0f;
}

__global__ void k_deg(const int* __restrict__ ei, float* __restrict__ deg, int E) {
    int e = blockIdx.x * 256 + threadIdx.x;
    if (e < E) atomicAdd(&deg[ei[E + e]], 1.0f);   // dst row of edge_index
}

__global__ void k_dinv(float* __restrict__ deg, int n) {
    int i = blockIdx.x * 256 + threadIdx.x;
    if (i < n) deg[i] = rsqrtf(deg[i] + 1.0f);     // +1 self loop
}

// h[i,:] = m[i,:]*dinv[i]^2 + b   (self-loop term + bias init)
__global__ void k_selfbias(const float* __restrict__ m, const float* __restrict__ dinv,
                           const float* __restrict__ b, float* __restrict__ h,
                           int Nn, int F) {
    int idx = blockIdx.x * 256 + threadIdx.x;
    if (idx < Nn * F) {
        int i = idx / F, f = idx - i * F;
        float d = dinv[i];
        h[idx] = m[idx] * d * d + b[f];
    }
}

// one wave per edge: h[dst,:] += m[src,:] * dinv[src]*dinv[dst]
__global__ __launch_bounds__(256) void k_scatter(const int* __restrict__ ei,
                                                 const float* __restrict__ dinv,
                                                 const float* __restrict__ m,
                                                 float* __restrict__ h,
                                                 int E, int F) {
    int gw   = (blockIdx.x * 256 + threadIdx.x) >> 5;
    int lane = threadIdx.x & 31;
    if (gw >= E) return;
    int s = ei[gw], d = ei[E + gw];
    float nrm = dinv[s] * dinv[d];
    const float* ms = m + (size_t)s * F;
    float*       hd = h + (size_t)d * F;
    for (int f = lane; f < F; f += 32)
        atomicAdd(&hd[f], ms[f] * nrm);
}

// one wave per row: LayerNorm (+ optional LeakyReLU), in place
__global__ __launch_bounds__(256) void k_ln(float* __restrict__ h,
                                            const float* __restrict__ g,
                                            const float* __restrict__ be,
                                            int Nn, int F, int lrelu) {
    int gw   = (blockIdx.x * 256 + threadIdx.x) >> 5;
    int lane = threadIdx.x & 31;
    if (gw >= Nn) return;
    float* row = h + (size_t)gw * F;
    float s = 0.0f, sq = 0.0f;
    for (int f = lane; f < F; f += 32) { float v = row[f]; s += v; sq += v * v; }
    #pragma unroll
    for (int o = 16; o > 0; o >>= 1) {
        s  += __shfl_xor(s,  o, 32);
        sq += __shfl_xor(sq, o, 32);
    }
    float mu  = s / (float)F;
    float var = sq / (float)F - mu * mu;
    float rs  = rsqrtf(var + 1e-5f);
    for (int f = lane; f < F; f += 32) {
        float y = (row[f] - mu) * rs * g[f] + be[f];
        if (lrelu) y = (y > 0.0f) ? y : 0.01f * y;
        row[f] = y;
    }
}

// one wave per node: sums[batch[i],:] += h[i,:] ; cnt[batch[i]] += 1
__global__ __launch_bounds__(256) void k_pool(const float* __restrict__ h,
                                              const int* __restrict__ batch,
                                              float* __restrict__ sums,
                                              float* __restrict__ cnt,
                                              int Nn, int F) {
    int gw   = (blockIdx.x * 256 + threadIdx.x) >> 5;
    int lane = threadIdx.x & 31;
    if (gw >= Nn) return;
    int gi = batch[gw];
    const float* row = h    + (size_t)gw * F;
    float*      srow = sums + (size_t)gi * F;
    for (int f = lane; f < F; f += 32)
        atomicAdd(&srow[f], row[f]);
    if (lane == 0) atomicAdd(&cnt[gi], 1.0f);
}

__global__ void k_pooldiv(const float* __restrict__ sums, const float* __restrict__ cnt,
                          float* __restrict__ pooled, int G, int F) {
    int idx = blockIdx.x * 256 + threadIdx.x;
    if (idx < G * F) {
        int gi = idx / F;
        pooled[idx] = sums[idx] / fmaxf(cnt[gi], 1.0f);
    }
}

// ---------------------------------------------------------------------------
static inline int cdiv(int a, int b) { return (a + b - 1) / b; }

extern "C" void kernel_launch(void* const* d_in, const int* in_sizes, int n_in,
                              void* d_out, int out_size, void* d_ws, size_t ws_size,
                              hipStream_t stream) {
    (void)n_in; (void)out_size; (void)ws_size;
    // inputs in setup_inputs() order
    const float* x     = (const float*)d_in[0];
    const int*   ei    = (const int*)  d_in[1];
    const int*   batch = (const int*)  d_in[2];
    const float* W1 = (const float*)d_in[4];  const float* b1 = (const float*)d_in[5];
    const float* g1 = (const float*)d_in[6];  const float* be1= (const float*)d_in[7];
    const float* W2 = (const float*)d_in[8];  const float* b2 = (const float*)d_in[9];
    const float* g2 = (const float*)d_in[10]; const float* be2= (const float*)d_in[11];
    const float* W3 = (const float*)d_in[12]; const float* b3 = (const float*)d_in[13];
    const float* g3 = (const float*)d_in[14]; const float* be3= (const float*)d_in[15];
    const float* Wf = (const float*)d_in[16]; const float* bf = (const float*)d_in[17];

    const int d2   = in_sizes[5];          // 256
    const int d4   = in_sizes[9];          // 512
    const int DOUT = in_sizes[17];         // 128
    const int DIN  = in_sizes[4] / d2;     // 128
    const int Nn   = in_sizes[0] / DIN;    // 50000
    const int E    = in_sizes[1] / 2;      // 800000
    const int G    = 128;                  // N_GRAPHS (fixed by reference)

    // workspace carve-up (float units)
    float* ws = (float*)d_ws;
    size_t o = 0;
    float* dinv   = ws + o; o += (size_t)((Nn + 63) & ~63);
    float* mbuf   = ws + o; o += (size_t)Nn * d4;
    float* hbuf   = ws + o; o += (size_t)Nn * d4;
    float* sums   = ws + o; o += (size_t)G * d4;
    float* pooled = ws + o; o += (size_t)G * d4;
    float* cnt    = ws + o; o += (size_t)((G + 63) & ~63);
    const size_t wmax = (size_t)d4 * d4;             // max weight elements
    __bf16* wth = (__bf16*)(ws + o); o += wmax / 2;  // bf16 hi plane
    __bf16* wtl = (__bf16*)(ws + o); o += wmax / 2;  // bf16 lo plane

    const dim3 B(256);

    // degree / normalization
    k_zero<<<cdiv(Nn, 256), B, 0, stream>>>(dinv, Nn);
    k_zero<<<cdiv(G * d4, 256), B, 0, stream>>>(sums, G * d4);
    k_zero<<<1, B, 0, stream>>>(cnt, G);
    k_deg <<<cdiv(E, 256), B, 0, stream>>>(ei, dinv, E);
    k_dinv<<<cdiv(Nn, 256), B, 0, stream>>>(dinv, Nn);

    // ---- layer 1: 128 -> 256 ----
    k_wprep<<<cdiv(DIN * d2, 256), B, 0, stream>>>(W1, wth, wtl, DIN, d2);
    k_gemm_wmma<<<dim3(cdiv(Nn, 64), d2 / 64), B, 0, stream>>>(x, wth, wtl, nullptr, mbuf, Nn, DIN, d2);
    k_selfbias<<<cdiv(Nn * d2, 256), B, 0, stream>>>(mbuf, dinv, b1, hbuf, Nn, d2);
    k_scatter <<<cdiv(E, 8), B, 0, stream>>>(ei, dinv, mbuf, hbuf, E, d2);
    k_ln      <<<cdiv(Nn, 8), B, 0, stream>>>(hbuf, g1, be1, Nn, d2, 1);

    // ---- layer 2: 256 -> 512 ----
    k_wprep<<<cdiv(d2 * d4, 256), B, 0, stream>>>(W2, wth, wtl, d2, d4);
    k_gemm_wmma<<<dim3(cdiv(Nn, 64), d4 / 64), B, 0, stream>>>(hbuf, wth, wtl, nullptr, mbuf, Nn, d2, d4);
    k_selfbias<<<cdiv(Nn * d4, 256), B, 0, stream>>>(mbuf, dinv, b2, hbuf, Nn, d4);
    k_scatter <<<cdiv(E, 8), B, 0, stream>>>(ei, dinv, mbuf, hbuf, E, d4);
    k_ln      <<<cdiv(Nn, 8), B, 0, stream>>>(hbuf, g2, be2, Nn, d4, 1);

    // ---- layer 3: 512 -> 512 ----
    k_wprep<<<cdiv(d4 * d4, 256), B, 0, stream>>>(W3, wth, wtl, d4, d4);
    k_gemm_wmma<<<dim3(cdiv(Nn, 64), d4 / 64), B, 0, stream>>>(hbuf, wth, wtl, nullptr, mbuf, Nn, d4, d4);
    k_selfbias<<<cdiv(Nn * d4, 256), B, 0, stream>>>(mbuf, dinv, b3, hbuf, Nn, d4);
    k_scatter <<<cdiv(E, 8), B, 0, stream>>>(ei, dinv, mbuf, hbuf, E, d4);
    k_ln      <<<cdiv(Nn, 8), B, 0, stream>>>(hbuf, g3, be3, Nn, d4, 0);  // no LeakyReLU

    // ---- global mean pool ----
    k_pool   <<<cdiv(Nn, 8), B, 0, stream>>>(hbuf, batch, sums, cnt, Nn, d4);
    k_pooldiv<<<cdiv(G * d4, 256), B, 0, stream>>>(sums, cnt, pooled, G, d4);

    // ---- final linear: pooled[G,512] @ Wf + bf -> d_out[G,128] ----
    k_wprep<<<cdiv(d4 * DOUT, 256), B, 0, stream>>>(Wf, wth, wtl, d4, DOUT);
    k_gemm_wmma<<<dim3(cdiv(G, 64), DOUT / 64), B, 0, stream>>>(pooled, wth, wtl, bf, (float*)d_out, G, d4, DOUT);
}
